// GT_85753317032541
// MI455X (gfx1250) — matrix operand order repaired and checked by reading, not compiled, not run on hardware
//
#include <hip/hip_runtime.h>
#include <math.h>

// ---------------- problem constants (match reference) ----------------
#define NN     30000
#define EE     300000
#define HH     128
#define NHEADS 8
#define DKV    16
#define LL     3
#define OUTD   64
#define GG     128
#define SPMAXV 64
#define NATOMV 9
#define AMAXV  119

typedef __bf16 bf16_t;
typedef bf16_t v16bf __attribute__((ext_vector_type(16)));
typedef bf16_t v8bf  __attribute__((ext_vector_type(8)));
typedef float  v8f   __attribute__((ext_vector_type(8)));

// ---------------- helpers ----------------
__device__ __forceinline__ float gelu_exact(float x) {
    return 0.5f * x * (1.0f + erff(x * 0.70710678118654752440f));
}

// signed float atomic max (classic int/uint trick; init value must be -inf)
__device__ __forceinline__ void atomicMaxFloat(float* addr, float val) {
    if (val >= 0.0f) atomicMax((int*)addr, __float_as_int(val));
    else             atomicMin((unsigned int*)addr, __float_as_uint(val));
}

// ---------------- generic fill ----------------
__global__ void fill_f32(float* __restrict__ p, float v, int n) {
    int i = blockIdx.x * blockDim.x + threadIdx.x;
    if (i < n) p[i] = v;
}

// ---------------- node embedding sum ----------------
__global__ __launch_bounds__(128)
void node_emb_kernel(const int* __restrict__ node_attr,
                     const float* __restrict__ atom_emb,
                     float* __restrict__ node_rep) {
    const int n = blockIdx.x;
    const int c = threadIdx.x;
    float s = 0.0f;
#pragma unroll
    for (int a = 0; a < NATOMV; ++a) {
        const int t = node_attr[n * NATOMV + a];
        s += atom_emb[((size_t)(a * AMAXV + t)) * HH + c];
    }
    node_rep[(size_t)n * HH + c] = s;
}

// ---------------- weight packing: f32 [128 x P] -> bf16 WMMA-B-fragment order --
// flat index t = ((ct*4 + kc)*32 + lane)*16 + j
//   maps element  k = kc*32 + (lane>>4)*16 + j,  col = ct*16 + (lane&15)
__global__ void pack_w_kernel(const float* __restrict__ W, bf16_t* __restrict__ out, int P) {
    const int t = blockIdx.x * blockDim.x + threadIdx.x;
    if (t >= P * 128) return;
    const int j    = t & 15;
    const int lane = (t >> 4) & 31;
    const int kc   = (t >> 9) & 3;
    const int ct   = t >> 11;
    const int k    = kc * 32 + (lane >> 4) * 16 + j;
    const int col  = ct * 16 + (lane & 15);
    out[t] = (bf16_t)W[(size_t)k * P + col];
}

// ---------------- WMMA GEMM: C[MxP] = preop(A)[Mx128] @ W[128xP] (+bias)(+R) ----
// PRE: 0 = identity, 1 = exact gelu (applied once while staging A into LDS)
// EPI: 0 = none, 1 = +bias[col], 2 = +bias[col] + R[row][col]
// P  : compile-time column count (128 or 64) -> all addressing becomes
//      one 64-bit base per lane + immediate offsets.
// Block: 256 threads = 8 waves; 64-row M-tile staged to LDS as bf16.
// Wave w owns cols [16w,16w+16): loads all 4 k-chunk B fragments ONCE (packed
// bf16, contiguous 32B/lane), then runs 4 row-subtiles x 4 WMMA reusing them.
template <int PRE, int EPI, int P>
__global__ __launch_bounds__(256)
void gemm128_wmma(const float* __restrict__ A, const bf16_t* __restrict__ Wp,
                  const float* __restrict__ bias, const float* __restrict__ R,
                  float* __restrict__ C, int M) {
    __shared__ bf16_t ldsA[64 * 128];
    const int tid   = threadIdx.x;
    const int mbase = blockIdx.x * 64;

    // stage 64x128 A tile: f32 -> (gelu) -> bf16, one ds_store_b128 per iter
    union U8 { bf16_t h[8]; uint4 u; };
    const float* abase = A + (size_t)mbase * 128 + tid * 8;
#pragma unroll
    for (int it = 0; it < 4; ++it) {
        const float4* ap = (const float4*)(abase + it * 2048);
        float4 f0 = ap[0], f1 = ap[1];
        if (PRE == 1) {
            f0.x = gelu_exact(f0.x); f0.y = gelu_exact(f0.y);
            f0.z = gelu_exact(f0.z); f0.w = gelu_exact(f0.w);
            f1.x = gelu_exact(f1.x); f1.y = gelu_exact(f1.y);
            f1.z = gelu_exact(f1.z); f1.w = gelu_exact(f1.w);
        }
        U8 u;
        u.h[0] = (bf16_t)f0.x; u.h[1] = (bf16_t)f0.y;
        u.h[2] = (bf16_t)f0.z; u.h[3] = (bf16_t)f0.w;
        u.h[4] = (bf16_t)f1.x; u.h[5] = (bf16_t)f1.y;
        u.h[6] = (bf16_t)f1.z; u.h[7] = (bf16_t)f1.w;
        ((uint4*)ldsA)[it * 256 + tid] = u.u;
    }
    __syncthreads();

    const int wave  = tid >> 5;
    const int cbase = wave * 16;
    if (cbase >= P) return;                 // uniform per-wave

    const int lane = tid & 31;
    const int half = lane >> 4;
    const int r16  = lane & 15;
    const int gcol = cbase + r16;

    // B fragments: packed, contiguous 32B per lane per chunk (immediate offsets)
    const bf16_t* wbase = Wp + wave * 2048 + lane * 16;
    v16bf bfrag[4];
#pragma unroll
    for (int kc = 0; kc < 4; ++kc)
        bfrag[kc] = *(const v16bf*)(wbase + kc * 512);

    const float binit = (EPI >= 1) ? bias[gcol] : 0.0f;

    // single 64-bit base per lane; all C/R accesses use immediate offsets
    float*       Cbase = C + (size_t)(mbase + half * 8) * P + gcol;
    const float* Rbase = (EPI == 2) ? (R + (size_t)(mbase + half * 8) * P + gcol) : nullptr;
    const bf16_t* lrow = &ldsA[r16 * 128 + half * 16];

#pragma unroll
    for (int sub = 0; sub < 4; ++sub) {
        if (mbase + sub * 16 >= M) break;   // 16-row granularity (M % 16 == 0)

        v8f acc;
#pragma unroll
        for (int j = 0; j < 8; ++j) acc[j] = binit;

#pragma unroll
        for (int kc = 0; kc < 4; ++kc) {
            // A fragment (contiguous k, matches packed B): lanes0-15 K={0..15},
            // lanes16-31 K={16..31} -> two 16B LDS loads (immediate offsets)
            const bf16_t* apL = lrow + sub * 16 * 128 + kc * 32;
            v8bf lo = *(const v8bf*)apL;
            v8bf hi = *(const v8bf*)(apL + 8);
            v16bf a = __builtin_shufflevector(lo, hi, 0, 1, 2, 3, 4, 5, 6, 7,
                                              8, 9, 10, 11, 12, 13, 14, 15);
            acc = __builtin_amdgcn_wmma_f32_16x16x32_bf16(
                false, a, false, bfrag[kc], (short)0, acc, false, false);
        }

        // C/D layout: VGPR j -> (lanes0-15: M=j, lanes16-31: M=j+8), N = lane%16
#pragma unroll
        for (int j = 0; j < 8; ++j) {
            const int off = (sub * 16 + j) * P;   // compile-time constant
            float v = acc[j];
            if (EPI == 2) v += Rbase[off];
            Cbase[off] = v;
        }
    }
}

// ---------------- attention edge kernels ----------------
__global__ void score_kernel(const float* __restrict__ Qn, const float* __restrict__ Kn,
                             const float* __restrict__ spK,
                             const int* __restrict__ ei, const int* __restrict__ sp,
                             float* __restrict__ score, float* __restrict__ mbuf) {
    const int t = blockIdx.x * blockDim.x + threadIdx.x;
    if (t >= EE * NHEADS) return;
    const int e = t >> 3, h = t & 7;
    const int src = ei[e], dst = ei[EE + e], spe = sp[e];
    const float* q  = Qn  + (size_t)dst * HH + h * DKV;
    const float* k  = Kn  + (size_t)src * HH + h * DKV;
    const float* ks = spK + (size_t)spe * HH + h * DKV;
    float s = 0.0f;
#pragma unroll
    for (int d = 0; d < DKV; ++d) s += q[d] * (k[d] + ks[d]);
    s *= 0.25f;                                  // 1/sqrt(16)
    score[t] = s;
    atomicMaxFloat(&mbuf[(size_t)dst * NHEADS + h], s);
}

__global__ void expsum_kernel(const int* __restrict__ ei,
                              float* __restrict__ score,
                              const float* __restrict__ mbuf,
                              float* __restrict__ sbuf) {
    const int t = blockIdx.x * blockDim.x + threadIdx.x;
    if (t >= EE * NHEADS) return;
    const int e = t >> 3, h = t & 7;
    const int dst = ei[EE + e];
    const float ev = __expf(score[t] - mbuf[(size_t)dst * NHEADS + h]);
    score[t] = ev;
    atomicAdd(&sbuf[(size_t)dst * NHEADS + h], ev);
}

__global__ void scatter_kernel(const float* __restrict__ Vn, const float* __restrict__ spV,
                               const int* __restrict__ ei, const int* __restrict__ sp,
                               const float* __restrict__ score, const float* __restrict__ sbuf,
                               float* __restrict__ aggr) {
    const long long t = (long long)blockIdx.x * blockDim.x + threadIdx.x;
    if (t >= (long long)EE * HH) return;
    const int e = (int)(t >> 7), c = (int)(t & 127);
    const int h = c >> 4;
    const int src = ei[e], dst = ei[EE + e], spe = sp[e];
    const float att = score[(size_t)e * NHEADS + h] /
                      (sbuf[(size_t)dst * NHEADS + h] + 1e-16f);
    const float v = (Vn[(size_t)src * HH + c] + spV[(size_t)spe * HH + c]) * att;
    atomicAdd(&aggr[(size_t)dst * HH + c], v);
}

// ---------------- layernorm: one wave32 per row ----------------
__global__ __launch_bounds__(256)
void layernorm_kernel(const float* __restrict__ X, const float* __restrict__ gamma,
                      const float* __restrict__ beta, float* __restrict__ Y, int M) {
    const int row = blockIdx.x * 8 + (threadIdx.x >> 5);
    if (row >= M) return;
    const int lane = threadIdx.x & 31;
    const float4 v = ((const float4*)(X + (size_t)row * HH))[lane];
    float s = v.x + v.y + v.z + v.w;
#pragma unroll
    for (int o = 16; o > 0; o >>= 1) s += __shfl_xor(s, o, 32);
    const float mu = s * (1.0f / 128.0f);
    const float dx = v.x - mu, dy = v.y - mu, dz = v.z - mu, dw = v.w - mu;
    float s2 = dx * dx + dy * dy + dz * dz + dw * dw;
#pragma unroll
    for (int o = 16; o > 0; o >>= 1) s2 += __shfl_xor(s2, o, 32);
    const float rs = rsqrtf(s2 * (1.0f / 128.0f) + 1e-5f);
    const int c = lane * 4;
    float4 o4;
    o4.x = dx * rs * gamma[c + 0] + beta[c + 0];
    o4.y = dy * rs * gamma[c + 1] + beta[c + 1];
    o4.z = dz * rs * gamma[c + 2] + beta[c + 2];
    o4.w = dw * rs * gamma[c + 3] + beta[c + 3];
    ((float4*)(Y + (size_t)row * HH))[lane] = o4;
}

// ---------------- pooling ----------------
__global__ void pool_cnt_kernel(const int* __restrict__ batch, float* __restrict__ cnt) {
    const int n = blockIdx.x * blockDim.x + threadIdx.x;
    if (n < NN) atomicAdd(&cnt[batch[n]], 1.0f);
}
__global__ void pool_sum_kernel(const float* __restrict__ node_rep,
                                const int* __restrict__ batch, float* __restrict__ pooled) {
    const int t = blockIdx.x * blockDim.x + threadIdx.x;
    if (t >= NN * HH) return;
    const int n = t >> 7, c = t & 127;
    atomicAdd(&pooled[(size_t)batch[n] * HH + c], node_rep[(size_t)n * HH + c]);
}
__global__ void pool_div_kernel(float* __restrict__ pooled, const float* __restrict__ cnt) {
    const int t = blockIdx.x * blockDim.x + threadIdx.x;
    if (t >= GG * HH) return;
    pooled[t] /= fmaxf(cnt[t >> 7], 1.0f);
}

// ---------------- host-side orchestration ----------------
extern "C" void kernel_launch(void* const* d_in, const int* in_sizes, int n_in,
                              void* d_out, int out_size, void* d_ws, size_t ws_size,
                              hipStream_t stream) {
    const int*   node_attr = (const int*)d_in[0];
    const int*   batch_idx = (const int*)d_in[1];
    const int*   edge_idx  = (const int*)d_in[2];
    const int*   sp        = (const int*)d_in[3];
    const float* atom_emb  = (const float*)d_in[4];
    const float* sp_emb    = (const float*)d_in[5];
    const float* lin_W     = (const float*)d_in[6];
    const float* lin_b     = (const float*)d_in[7];
    const float* sp_W      = (const float*)d_in[8];
    const float* norm1     = (const float*)d_in[9];
    const float* out_W     = (const float*)d_in[10];
    const float* out_b     = (const float*)d_in[11];
    const float* norm2     = (const float*)d_in[12];
    const float* head_W    = (const float*)d_in[13];
    const float* head_b    = (const float*)d_in[14];
    float* out = (float*)d_out;

    // workspace layout (floats)
    const size_t NHf = (size_t)NN * HH;
    float* ws       = (float*)d_ws;
    float* node_rep = ws;
    float* Qn       = node_rep + NHf;   // also FFN tmp1
    float* Kn       = Qn + NHf;         // also trans
    float* Vn       = Kn + NHf;         // also FFN tmp2
    float* aggr     = Vn + NHf;
    float* spT      = aggr + NHf;                    // 64x128
    float* spK      = spT + (size_t)SPMAXV * HH;
    float* spV      = spK + (size_t)SPMAXV * HH;
    float* score    = spV + (size_t)SPMAXV * HH;     // E x 8
    float* mbuf     = score + (size_t)EE * NHEADS;   // N x 8
    float* sbuf     = mbuf + (size_t)NN * NHEADS;    // N x 8
    float* pooled   = sbuf + (size_t)NN * NHEADS;    // G x 128
    float* cnt      = pooled + (size_t)GG * HH;      // G

    // packed bf16 weights (WMMA B-fragment order)
    const int WMAT = 128 * 128;                      // elems per 128-col matrix
    bf16_t* pwLin = (bf16_t*)(cnt + GG);             // 12 * WMAT
    bf16_t* pwSp  = pwLin + (size_t)12 * WMAT;       // 3 * WMAT
    bf16_t* pwO   = pwSp + (size_t)3 * WMAT;         // 3 * WMAT
    bf16_t* pwH   = pwO + (size_t)3 * WMAT;          // 128*64

    const int HH2 = HH * HH;
    const dim3 B256(256);

    // pack all weights once per call
    for (int i = 0; i < 12; ++i)
        pack_w_kernel<<<(WMAT + 255) / 256, B256, 0, stream>>>(
            lin_W + (size_t)i * HH2, pwLin + (size_t)i * WMAT, HH);
    for (int l = 0; l < LL; ++l) {
        pack_w_kernel<<<(WMAT + 255) / 256, B256, 0, stream>>>(
            sp_W + (size_t)l * HH2, pwSp + (size_t)l * WMAT, HH);
        pack_w_kernel<<<(WMAT + 255) / 256, B256, 0, stream>>>(
            out_W + (size_t)l * HH2, pwO + (size_t)l * WMAT, HH);
    }
    pack_w_kernel<<<(128 * OUTD + 255) / 256, B256, 0, stream>>>(head_W, pwH, OUTD);

    // node embedding
    node_emb_kernel<<<NN, 128, 0, stream>>>(node_attr, atom_emb, node_rep);

    const int GRID_N  = (NN + 63) / 64;   // 469 (last block: 3 of 4 subtiles)
    const int GRID_SP = 1;                // 64 rows exactly
    const int GRID_G  = GG / 64;          // 2

    for (int l = 0; l < LL; ++l) {
        bf16_t* qWp = pwLin + (size_t)(l * 4 + 0) * WMAT;
        bf16_t* kWp = pwLin + (size_t)(l * 4 + 1) * WMAT;
        bf16_t* vWp = pwLin + (size_t)(l * 4 + 2) * WMAT;
        bf16_t* aWp = pwLin + (size_t)(l * 4 + 3) * WMAT;
        const float* qb = lin_b + (size_t)(l * 4 + 0) * HH;
        const float* kb = lin_b + (size_t)(l * 4 + 1) * HH;
        const float* vb = lin_b + (size_t)(l * 4 + 2) * HH;
        const float* ab = lin_b + (size_t)(l * 4 + 3) * HH;
        const float* obl = out_b + (size_t)l * HH;
        const float* g1 = norm1 + (size_t)l * 2 * HH;
        const float* b1 = g1 + HH;
        const float* g2 = norm2 + (size_t)l * 2 * HH;
        const float* b2 = g2 + HH;

        // shortest-path tables
        gemm128_wmma<0, 0, HH><<<GRID_SP, B256, 0, stream>>>(
            sp_emb, pwSp + (size_t)l * WMAT, nullptr, nullptr, spT, SPMAXV);
        gemm128_wmma<0, 0, HH><<<GRID_SP, B256, 0, stream>>>(spT, kWp, nullptr, nullptr, spK, SPMAXV);
        gemm128_wmma<0, 0, HH><<<GRID_SP, B256, 0, stream>>>(spT, vWp, nullptr, nullptr, spV, SPMAXV);

        // node-level projections
        gemm128_wmma<0, 1, HH><<<GRID_N, B256, 0, stream>>>(node_rep, qWp, qb, nullptr, Qn, NN);
        gemm128_wmma<0, 1, HH><<<GRID_N, B256, 0, stream>>>(node_rep, kWp, kb, nullptr, Kn, NN);
        gemm128_wmma<0, 1, HH><<<GRID_N, B256, 0, stream>>>(node_rep, vWp, vb, nullptr, Vn, NN);

        // segment softmax state
        fill_f32<<<(NN * NHEADS + 255) / 256, B256, 0, stream>>>(mbuf, -INFINITY, NN * NHEADS);
        fill_f32<<<(NN * NHEADS + 255) / 256, B256, 0, stream>>>(sbuf, 0.0f, NN * NHEADS);
        fill_f32<<<(int)((NHf + 255) / 256), B256, 0, stream>>>(aggr, 0.0f, (int)NHf);

        score_kernel<<<(EE * NHEADS + 255) / 256, B256, 0, stream>>>(Qn, Kn, spK, edge_idx, sp, score, mbuf);
        expsum_kernel<<<(EE * NHEADS + 255) / 256, B256, 0, stream>>>(edge_idx, score, mbuf, sbuf);
        scatter_kernel<<<(int)(((long long)EE * HH + 255) / 256), B256, 0, stream>>>(
            Vn, spV, edge_idx, sp, score, sbuf, aggr);

        // tmp1 = gelu(aggr)@aW + ab + node_rep  (into Qn)
        gemm128_wmma<1, 2, HH><<<GRID_N, B256, 0, stream>>>(aggr, aWp, ab, node_rep, Qn, NN);
        // trans = LN(tmp1)  (into Kn)
        layernorm_kernel<<<NN / 8, B256, 0, stream>>>(Qn, g1, b1, Kn, NN);
        // tmp2 = gelu(trans)@out_W + out_b + trans  (into Vn)
        gemm128_wmma<1, 2, HH><<<GRID_N, B256, 0, stream>>>(
            Kn, pwO + (size_t)l * WMAT, obl, Kn, Vn, NN);
        // node_rep = LN(tmp2)
        layernorm_kernel<<<NN / 8, B256, 0, stream>>>(Vn, g2, b2, node_rep, NN);
    }

    // mean pooling + head
    fill_f32<<<(GG * HH + GG + 255) / 256, B256, 0, stream>>>(pooled, 0.0f, GG * HH + GG);
    pool_cnt_kernel<<<(NN + 255) / 256, B256, 0, stream>>>(batch_idx, cnt);
    pool_sum_kernel<<<(NN * HH + 255) / 256, B256, 0, stream>>>(node_rep, batch_idx, pooled);
    pool_div_kernel<<<(GG * HH + 255) / 256, B256, 0, stream>>>(pooled, cnt);
    gemm128_wmma<0, 1, OUTD><<<GRID_G, B256, 0, stream>>>(pooled, pwH, head_b, nullptr, out, GG);

    (void)in_sizes; (void)n_in; (void)out_size; (void)ws_size;
}
